// SelfAttention_87265145520355
// MI455X (gfx1250) — matrix operand order repaired and checked by reading, not compiled, not run on hardware
//
#include <hip/hip_runtime.h>

// Self-attention forward, MI455X (gfx1250), wave32 + WMMA f16->f32.
// Workspace use: ~48 MB (f16 copies of x/weights + Q/K/Vt/ctx buffers).

typedef __attribute__((ext_vector_type(16))) _Float16 v16h;
typedef __attribute__((ext_vector_type(8)))  _Float16 v8h;
typedef __attribute__((ext_vector_type(8)))  float    v8f;

#define D_MODEL 1024
#define SEQ     2048
#define NH      16
#define DH      64

__device__ __forceinline__ v8f wmma16(v16h a, v16h b, v8f c) {
    return __builtin_amdgcn_wmma_f32_16x16x32_f16(
        /*neg_a=*/false, a, /*neg_b=*/false, b,
        /*c_mod=*/(short)0, c, /*reuse_a=*/false, /*reuse_b=*/false);
}

// A-operand (16 rows x 32 K, f16). ISA 7.12.2: lane L holds row M=L&15;
// lanes 0-15: K = 0..7 (V0..3) and 16..23 (V4..7); lanes 16-31: K = 8..15, 24..31.
__device__ __forceinline__ v16h load_a_frag(const _Float16* p, int ld) {
    int lane = threadIdx.x & 31;
    int row  = lane & 15;
    int k0   = (lane >> 4) * 8;
    union { v16h v; v8h h[2]; } u;
    u.h[0] = *(const v8h*)(p + (size_t)row * ld + k0);
    u.h[1] = *(const v8h*)(p + (size_t)row * ld + k0 + 16);
    return u.v;
}

// B-operand (32 K x 16 cols, f16) from memory laid out [col][K] (row-major W).
// Lane L holds col N=L&15; lanes 0-15: K=0..7,16..23, lanes 16-31: K=8..15,24..31.
__device__ __forceinline__ v16h load_b_frag(const _Float16* p, int ld) {
    int lane = threadIdx.x & 31;
    int n    = lane & 15;
    int k0   = (lane >> 4) * 8;
    union { v16h v; v8h h[2]; } u;
    u.h[0] = *(const v8h*)(p + (size_t)n * ld + k0);
    u.h[1] = *(const v8h*)(p + (size_t)n * ld + k0 + 16);
    return u.v;
}

// ---------------------------------------------------------------- convert
__global__ void __launch_bounds__(256)
cvt_f32_to_f16_x4(const float* __restrict__ src, _Float16* __restrict__ dst, int n4) {
    int i = blockIdx.x * blockDim.x + threadIdx.x;
    if (i < n4) {
        float4 f = ((const float4*)src)[i];
        v8h* d = nullptr; (void)d;
        _Float16 h0 = (_Float16)f.x, h1 = (_Float16)f.y,
                 h2 = (_Float16)f.z, h3 = (_Float16)f.w;
        _Float16* o = dst + (size_t)i * 4;
        o[0] = h0; o[1] = h1; o[2] = h2; o[3] = h3;
    }
}

// ---------------------------------------------------------------- QKV GEMM
// Y = xh @ W^T + b ; per-wave 32x32 tile, block = 8 waves = 64 x 128 tile.
// which (blockIdx.z): 0=Q (scaled 1/8, layout [B,H,S,64]),
//                     1=K (layout [B,H,S,64]), 2=V (transposed [B,H,64,S]).
__global__ void __launch_bounds__(256)
qkv_gemm_kernel(const _Float16* __restrict__ xh,
                const _Float16* __restrict__ wq, const _Float16* __restrict__ wk,
                const _Float16* __restrict__ wv,
                const float* __restrict__ bq, const float* __restrict__ bk,
                const float* __restrict__ bv,
                _Float16* __restrict__ Qb, _Float16* __restrict__ Kb,
                _Float16* __restrict__ Vt) {
    const int which = blockIdx.z;
    const _Float16* w   = (which == 0) ? wq : (which == 1) ? wk : wv;
    const float*    bia = (which == 0) ? bq : (which == 1) ? bk : bv;

    const int wave = threadIdx.x >> 5;
    const int lane = threadIdx.x & 31;
    const int mBase = blockIdx.x * 64  + (wave >> 2) * 32;
    const int nBase = blockIdx.y * 128 + (wave & 3) * 32;

    v8f c[2][2] = {};
    for (int k = 0; k < D_MODEL; k += 32) {
        __builtin_prefetch(w + (size_t)nBase * D_MODEL + k + 256, 0, 0);
        v16h a0 = load_a_frag(xh + (size_t)mBase       * D_MODEL + k, D_MODEL);
        v16h a1 = load_a_frag(xh + (size_t)(mBase + 16) * D_MODEL + k, D_MODEL);
        v16h b0 = load_b_frag(w  + (size_t)nBase       * D_MODEL + k, D_MODEL);
        v16h b1 = load_b_frag(w  + (size_t)(nBase + 16) * D_MODEL + k, D_MODEL);
        c[0][0] = wmma16(a0, b0, c[0][0]);
        c[0][1] = wmma16(a0, b1, c[0][1]);
        c[1][0] = wmma16(a1, b0, c[1][0]);
        c[1][1] = wmma16(a1, b1, c[1][1]);
    }

    const int half = lane >> 4, ln = lane & 15;
#pragma unroll
    for (int ni = 0; ni < 2; ++ni) {
        const int n = nBase + ni * 16 + ln;
        const float bn = bia[n];
        const int hh = n >> 6, dd = n & 63;
#pragma unroll
        for (int mi = 0; mi < 2; ++mi) {
#pragma unroll
            for (int j = 0; j < 8; ++j) {
                const int m = mBase + mi * 16 + half * 8 + j;
                float val = c[mi][ni][j] + bn;
                if (which == 0) val *= 0.125f;               // fold 1/sqrt(DH) into Q
                const int bb = m >> 11, ss = m & (SEQ - 1);
                if (which < 2) {
                    _Float16* dst = (which == 0) ? Qb : Kb;
                    dst[((((size_t)bb * NH + hh) * SEQ) + ss) * DH + dd] = (_Float16)val;
                } else {
                    Vt[((((size_t)bb * NH + hh) * DH) + dd) * SEQ + ss] = (_Float16)val;
                }
            }
        }
    }
}

// ---------------------------------------------------------------- attention
// Block: 128 threads (4 waves); wave owns 16 query rows; block = 64 queries
// of one (b,h). Online (flash) softmax over 64-key chunks, all matmuls WMMA.
__global__ void __launch_bounds__(128)
attention_kernel(const _Float16* __restrict__ Qb, const _Float16* __restrict__ Kb,
                 const _Float16* __restrict__ Vt, _Float16* __restrict__ ctxh) {
    __shared__ _Float16 lds_p[4][16 * 64];            // per-wave P tile (wave-private)

    const int wave = threadIdx.x >> 5;
    const int lane = threadIdx.x & 31;
    const int half = lane >> 4, ln = lane & 15;

    const int h  = blockIdx.y;
    const int b  = blockIdx.z;
    const int bh = b * NH + h;
    const int q0 = blockIdx.x * 64 + wave * 16;       // this wave's 16 query rows

    const _Float16* qp = Qb + ((size_t)bh * SEQ + q0) * DH;
    v16h aq0 = load_a_frag(qp,      DH);              // K-dim 0..31
    v16h aq1 = load_a_frag(qp + 32, DH);              // K-dim 32..63

    float run_max[8], run_sum[8];
#pragma unroll
    for (int j = 0; j < 8; ++j) { run_max[j] = -1e30f; run_sum[j] = 0.0f; }
    v8f o4[4] = {};

    _Float16* pl = &lds_p[wave][0];

    for (int kc = 0; kc < SEQ / 64; ++kc) {
        const int keyBase = kc * 64;

        // ---- scores: 16 x 64 tile = 4 accumulators
        v8f s4[4] = {};
#pragma unroll
        for (int nt = 0; nt < 4; ++nt) {
            const _Float16* kp = Kb + ((size_t)bh * SEQ + keyBase + nt * 16) * DH;
            v16h bk0 = load_b_frag(kp,      DH);
            v16h bk1 = load_b_frag(kp + 32, DH);
            s4[nt] = wmma16(aq0, bk0, s4[nt]);
            s4[nt] = wmma16(aq1, bk1, s4[nt]);
        }

        // ---- online softmax: row max (local + cross-lane within 16-lane half)
        float mnew[8];
#pragma unroll
        for (int j = 0; j < 8; ++j) {
            float m = s4[0][j];
            m = fmaxf(m, s4[1][j]); m = fmaxf(m, s4[2][j]); m = fmaxf(m, s4[3][j]);
#pragma unroll
            for (int mask = 1; mask <= 8; mask <<= 1)
                m = fmaxf(m, __shfl_xor(m, mask, 32));
            mnew[j] = m;
        }

        float alpha[8];
#pragma unroll
        for (int j = 0; j < 8; ++j) {
            float nm = fmaxf(run_max[j], mnew[j]);
            alpha[j] = __expf(run_max[j] - nm);
            run_max[j] = nm;
        }

        // ---- P = exp(s - max), row sums
        float psum[8];
#pragma unroll
        for (int j = 0; j < 8; ++j) psum[j] = 0.0f;
#pragma unroll
        for (int nt = 0; nt < 4; ++nt) {
#pragma unroll
            for (int j = 0; j < 8; ++j) {
                float p = __expf(s4[nt][j] - run_max[j]);
                s4[nt][j] = p;
                psum[j] += p;
            }
        }
#pragma unroll
        for (int j = 0; j < 8; ++j) {
#pragma unroll
            for (int mask = 1; mask <= 8; mask <<= 1)
                psum[j] += __shfl_xor(psum[j], mask, 32);
            run_sum[j] = run_sum[j] * alpha[j] + psum[j];
        }

        // ---- rescale running output
#pragma unroll
        for (int nt = 0; nt < 4; ++nt)
#pragma unroll
            for (int j = 0; j < 8; ++j) o4[nt][j] *= alpha[j];

        // ---- C-layout -> A-layout via wave-private LDS (DS ops in-order per wave)
#pragma unroll
        for (int nt = 0; nt < 4; ++nt)
#pragma unroll
            for (int j = 0; j < 8; ++j)
                pl[(j + half * 8) * 64 + nt * 16 + ln] = (_Float16)s4[nt][j];

        v16h ap0 = load_a_frag(pl,      64);          // keys 0..31 of chunk
        v16h ap1 = load_a_frag(pl + 32, 64);          // keys 32..63

        // ---- ctx += P @ V
#pragma unroll
        for (int nt = 0; nt < 4; ++nt) {
            const _Float16* vp = Vt + ((size_t)bh * DH + nt * 16) * SEQ + keyBase;
            v16h bv0 = load_b_frag(vp,      SEQ);
            v16h bv1 = load_b_frag(vp + 32, SEQ);
            o4[nt] = wmma16(ap0, bv0, o4[nt]);
            o4[nt] = wmma16(ap1, bv1, o4[nt]);
        }
    }

    // ---- normalize and emit ctx as [B,S,H,DH] (= [4096,1024] f16)
    float inv[8];
#pragma unroll
    for (int j = 0; j < 8; ++j) inv[j] = 1.0f / run_sum[j];
#pragma unroll
    for (int nt = 0; nt < 4; ++nt) {
#pragma unroll
        for (int j = 0; j < 8; ++j) {
            const int row = q0 + half * 8 + j;
            const int col = h * DH + nt * 16 + ln;
            ctxh[((size_t)b * SEQ + row) * D_MODEL + col] = (_Float16)(o4[nt][j] * inv[j]);
        }
    }
}

// ---------------------------------------------------------------- out proj
__global__ void __launch_bounds__(256)
out_gemm_kernel(const _Float16* __restrict__ ctxh, const _Float16* __restrict__ wo,
                const float* __restrict__ bo, float* __restrict__ out) {
    const int wave = threadIdx.x >> 5;
    const int lane = threadIdx.x & 31;
    const int mBase = blockIdx.x * 64  + (wave >> 2) * 32;
    const int nBase = blockIdx.y * 128 + (wave & 3) * 32;

    v8f c[2][2] = {};
    for (int k = 0; k < D_MODEL; k += 32) {
        __builtin_prefetch(wo + (size_t)nBase * D_MODEL + k + 256, 0, 0);
        v16h a0 = load_a_frag(ctxh + (size_t)mBase       * D_MODEL + k, D_MODEL);
        v16h a1 = load_a_frag(ctxh + (size_t)(mBase + 16) * D_MODEL + k, D_MODEL);
        v16h b0 = load_b_frag(wo   + (size_t)nBase       * D_MODEL + k, D_MODEL);
        v16h b1 = load_b_frag(wo   + (size_t)(nBase + 16) * D_MODEL + k, D_MODEL);
        c[0][0] = wmma16(a0, b0, c[0][0]);
        c[0][1] = wmma16(a0, b1, c[0][1]);
        c[1][0] = wmma16(a1, b0, c[1][0]);
        c[1][1] = wmma16(a1, b1, c[1][1]);
    }

    const int half = lane >> 4, ln = lane & 15;
#pragma unroll
    for (int ni = 0; ni < 2; ++ni) {
        const int n = nBase + ni * 16 + ln;
        const float bn = bo[n];
#pragma unroll
        for (int mi = 0; mi < 2; ++mi) {
#pragma unroll
            for (int j = 0; j < 8; ++j) {
                const int m = mBase + mi * 16 + half * 8 + j;
                out[(size_t)m * D_MODEL + n] = c[mi][ni][j] + bn;
            }
        }
    }
}

// ---------------------------------------------------------------- launch
extern "C" void kernel_launch(void* const* d_in, const int* in_sizes, int n_in,
                              void* d_out, int out_size, void* d_ws, size_t ws_size,
                              hipStream_t stream) {
    const float* x  = (const float*)d_in[0];
    const float* wq = (const float*)d_in[1];
    const float* bq = (const float*)d_in[2];
    const float* wk = (const float*)d_in[3];
    const float* bk = (const float*)d_in[4];
    const float* wv = (const float*)d_in[5];
    const float* bv = (const float*)d_in[6];
    const float* wo = (const float*)d_in[7];
    const float* bo = (const float*)d_in[8];
    float* out = (float*)d_out;

    const size_t NX = (size_t)2 * SEQ * D_MODEL;      // 4,194,304
    const size_t NW = (size_t)D_MODEL * D_MODEL;      // 1,048,576

    _Float16* xh   = (_Float16*)d_ws;
    _Float16* wqh  = xh  + NX;
    _Float16* wkh  = wqh + NW;
    _Float16* wvh  = wkh + NW;
    _Float16* woh  = wvh + NW;
    _Float16* Qb   = woh + NW;
    _Float16* Kb   = Qb  + NX;
    _Float16* Vt   = Kb  + NX;
    _Float16* ctxh = Vt  + NX;   // total ~48 MB of f16 workspace

    // 1) f32 -> f16 conversions
    cvt_f32_to_f16_x4<<<(int)(NX / 4 / 256), 256, 0, stream>>>(x,  xh,  (int)(NX / 4));
    cvt_f32_to_f16_x4<<<(int)(NW / 4 / 256), 256, 0, stream>>>(wq, wqh, (int)(NW / 4));
    cvt_f32_to_f16_x4<<<(int)(NW / 4 / 256), 256, 0, stream>>>(wk, wkh, (int)(NW / 4));
    cvt_f32_to_f16_x4<<<(int)(NW / 4 / 256), 256, 0, stream>>>(wv, wvh, (int)(NW / 4));
    cvt_f32_to_f16_x4<<<(int)(NW / 4 / 256), 256, 0, stream>>>(wo, woh, (int)(NW / 4));

    // 2) fused QKV projections (grid.z selects Q/K/V)
    qkv_gemm_kernel<<<dim3(64, 8, 3), 256, 0, stream>>>(
        xh, wqh, wkh, wvh, bq, bk, bv, Qb, Kb, Vt);

    // 3) flash attention
    attention_kernel<<<dim3(SEQ / 64, NH, 2), 128, 0, stream>>>(Qb, Kb, Vt, ctxh);

    // 4) output projection -> f32 result
    out_gemm_kernel<<<dim3(64, 8, 1), 256, 0, stream>>>(ctxh, woh, bo, out);
}
